// SwinTransformerBlock_50534585205179
// MI455X (gfx1250) — matrix operand order repaired and checked by reading, not compiled, not run on hardware
//
#include <hip/hip_runtime.h>
#include <hip/hip_bf16.h>
#include <cstdint>
#include <cstddef>

typedef __attribute__((ext_vector_type(16))) _Float16 v16h;
typedef __attribute__((ext_vector_type(8)))  _Float16 v8h;
typedef __attribute__((ext_vector_type(8)))  float    v8f;
typedef __attribute__((ext_vector_type(4)))  float    v4f;

#define L_    16384
#define C_    512
#define NH_   16
#define HD_   32
#define WSZ   128
#define SSZ   64
#define NW_   128
#define B_    2
#define HID_  2048
#define MTOK  (B_ * L_)          // 32768 token rows (shifted order)
#define LOGIT_MAX 4.6051702f     // ln(100)

// ---------------------------------------------------------------------------
// CDNA5 async copy: global f16 -> LDS, 16 bytes per lane, ASYNCcnt-tracked.
// Generic LDS pointers carry the wave-relative LDS offset in bits [31:0]
// (ISA 10.2 aperture mapping), which is exactly the VDST operand format.
// ---------------------------------------------------------------------------
__device__ __forceinline__ void async_copy_b128(const _Float16* gsrc, _Float16* ldst) {
    unsigned loff = (unsigned)(unsigned long long)(uintptr_t)ldst;
    unsigned long long ga = (unsigned long long)(uintptr_t)gsrc;
    asm volatile("global_load_async_to_lds_b128 %0, %1, off"
                 :: "v"(loff), "v"(ga) : "memory");
}
__device__ __forceinline__ void async_wait0() {
    asm volatile("s_wait_asynccnt 0" ::: "memory");
}

// ---------------------------------------------------------------------------
// WMMA fragment helpers (wave32 layouts per CDNA5 ISA 7.12.2)
// ---------------------------------------------------------------------------

// A matrix 16x32 f16, row-major source p[m*ld + k].
__device__ __forceinline__ v16h wm_load_a16(const _Float16* p, int ld, int lane) {
    int m   = lane & 15;
    int off = (lane & 16) ? 8 : 0;          // K sub-half per lane group
    const _Float16* row = p + m * ld;
    v16h a;
#pragma unroll
    for (int v = 0; v < 4; ++v) {
        a[2 * v]     = row[off + 2 * v];
        a[2 * v + 1] = row[off + 2 * v + 1];
        a[8 + 2 * v]     = row[16 + off + 2 * v];
        a[8 + 2 * v + 1] = row[16 + off + 2 * v + 1];
    }
    return a;
}

// A matrix 16x32 from an fp32 source (converted on load; used for P=softmax(S)).
__device__ __forceinline__ v16h wm_load_a32(const float* p, int ld, int lane) {
    int m   = lane & 15;
    int off = (lane & 16) ? 8 : 0;
    const float* row = p + m * ld;
    v16h a;
#pragma unroll
    for (int v = 0; v < 4; ++v) {
        a[2 * v]     = (_Float16)row[off + 2 * v];
        a[2 * v + 1] = (_Float16)row[off + 2 * v + 1];
        a[8 + 2 * v]     = (_Float16)row[16 + off + 2 * v];
        a[8 + 2 * v + 1] = (_Float16)row[16 + off + 2 * v + 1];
    }
    return a;
}

// B matrix 32(K)x16(N) where source is stored "transposed": p[n*ld + k].
__device__ __forceinline__ v16h wm_load_bT16(const _Float16* p, int ld, int lane) {
    int n  = lane & 15;
    int kh = (lane & 16) ? 8 : 0;
    const _Float16* row = p + n * ld;
    v16h b;
#pragma unroll
    for (int v = 0; v < 8; ++v) {
        b[v]     = row[kh + v];
        b[8 + v] = row[16 + kh + v];
    }
    return b;
}

__device__ __forceinline__ v8f wm_zero() {
    v8f z;
#pragma unroll
    for (int i = 0; i < 8; ++i) z[i] = 0.0f;
    return z;
}

__device__ __forceinline__ v8f wm_mma(v16h a, v16h b, v8f c) {
    return __builtin_amdgcn_wmma_f32_16x16x32_f16(false, a, false, b, (short)0, c,
                                                  false, false);
}

// ---------------------------------------------------------------------------
// Prep kernels
// ---------------------------------------------------------------------------

// Fused qkv bias: [q_bias, 0, v_bias] (1536)
__global__ void prep_qkvb(const float* __restrict__ qb, const float* __restrict__ vb,
                          float* __restrict__ out) {
    int i = blockIdx.x * 256 + threadIdx.x;
    if (i < 3 * C_) {
        float v = 0.0f;
        if (i < C_)             v = qb[i];
        else if (i >= 2 * C_)   v = vb[i - 2 * C_];
        out[i] = v;
    }
}

// SwinV2 continuous position bias table: biasT[255][NH] = 16*sigmoid(MLP(coord))
__global__ void prep_cpb(const float* __restrict__ w1, const float* __restrict__ b1,
                         const float* __restrict__ w2, float* __restrict__ biasT) {
    __shared__ float sh[512];
    int i = blockIdx.x;                        // relative index 0..254
    float t   = (float)(i - (WSZ - 1)) / (float)(WSZ - 1) * 8.0f;
    float sgn = (t > 0.f) ? 1.f : ((t < 0.f) ? -1.f : 0.f);
    float val = sgn * log2f(fabsf(t) + 1.0f) / 3.0f;   // log2(8)=3
    for (int j = threadIdx.x; j < 512; j += 256)
        sh[j] = fmaxf(val * w1[j] + b1[j], 0.0f);
    __syncthreads();
    if (threadIdx.x < NH_) {
        int h = threadIdx.x;
        float s = 0.0f;
        for (int j = 0; j < 512; ++j) s += sh[j] * w2[h * 512 + j];
        biasT[i * NH_ + h] = 16.0f / (1.0f + expf(-s));
    }
}

// roll(x, -SS) and cast fp32 -> fp16: Xs[b][t][c] = x[b][(t+SS)%L][c]
__global__ void prep_shift_cast(const float* __restrict__ x, _Float16* __restrict__ xs) {
    size_t gid  = (size_t)blockIdx.x * 256 + threadIdx.x;
    size_t base = gid * 8;
    int row = (int)(base >> 9);                // /512
    int c   = (int)(base & 511);
    int b = row >> 14, t = row & (L_ - 1);
    const float* src = x + ((size_t)(b << 14) + (size_t)((t + SSZ) & (L_ - 1))) * C_ + c;
    v4f f0 = *(const v4f*)src;
    v4f f1 = *(const v4f*)(src + 4);
    v8h h;
#pragma unroll
    for (int i = 0; i < 4; ++i) { h[i] = (_Float16)f0[i]; h[4 + i] = (_Float16)f1[i]; }
    *(v8h*)(xs + base) = h;
}

// ---------------------------------------------------------------------------
// WMMA GEMM:  Out[M,N](f16) = act( A[M,K](f16) @ W[N,K](f32)^T + bias[N] )
// Block tile 128x128, K step 32, 8 waves each computing 64x32 (4x2 tiles).
// A tile staged via GLOBAL_LOAD_ASYNC_TO_LDS_B128 (ASYNCcnt), W tile via
// VALU (fp32->f16 convert). `shift` rolls the stored row within a batch.
// ---------------------------------------------------------------------------
#define LDT 40   // padded LDS row stride (f16 elements)

template <int ACT>  // 0 = none, 1 = SiLU
__global__ __launch_bounds__(256) void gemm_f16(const _Float16* __restrict__ A,
                                                const float* __restrict__ W,
                                                const float* __restrict__ bias,
                                                _Float16* __restrict__ Out,
                                                int N, int K, int shift) {
    __shared__ _Float16 sA[2][128 * LDT];
    __shared__ _Float16 sW[2][128 * LDT];

    const int m0 = blockIdx.y * 128;
    const int n0 = blockIdx.x * 128;
    const int tid  = threadIdx.x;
    const int wave = tid >> 5, lane = tid & 31;
    const int wr = wave >> 2;      // 0..1 (row of 64)
    const int wc = wave & 3;       // 0..3 (col of 32)
    const int nk = K >> 5;

    auto stage = [&](int buf, int kk) {
        int r = tid >> 1;          // 0..127
        int h = tid & 1;           // half of the 32-wide K slab
        // A tile: 128 x 32 f16, async DMA straight into LDS (2 x 16B per thread)
        const _Float16* ga = A + (size_t)(m0 + r) * K + (kk + h * 16);
        async_copy_b128(ga,     &sA[buf][r * LDT + h * 16]);
        async_copy_b128(ga + 8, &sA[buf][r * LDT + h * 16 + 8]);
        // W tile: 128 x 32 fp32 -> f16 through VALU
        const float* gw = W + (size_t)(n0 + r) * K + (kk + h * 16);
        __builtin_prefetch(gw + 32, 0, 1);     // next K slab -> global_prefetch_b8
        v4f f0 = *(const v4f*)gw;
        v4f f1 = *(const v4f*)(gw + 4);
        v4f f2 = *(const v4f*)(gw + 8);
        v4f f3 = *(const v4f*)(gw + 12);
        v8h w0, w1;
#pragma unroll
        for (int i = 0; i < 4; ++i) {
            w0[i] = (_Float16)f0[i]; w0[4 + i] = (_Float16)f1[i];
            w1[i] = (_Float16)f2[i]; w1[4 + i] = (_Float16)f3[i];
        }
        *(v8h*)&sW[buf][r * LDT + h * 16]     = w0;
        *(v8h*)&sW[buf][r * LDT + h * 16 + 8] = w1;
    };

    v8f acc[4][2];
#pragma unroll
    for (int i = 0; i < 4; ++i)
#pragma unroll
        for (int j = 0; j < 2; ++j) acc[i][j] = wm_zero();

    stage(0, 0);
    async_wait0();
    __syncthreads();

    for (int kt = 0; kt < nk; ++kt) {
        int cur = kt & 1;
        if (kt + 1 < nk) stage(cur ^ 1, (kt + 1) * 32);   // async copy in flight...

        v16h af[4];
#pragma unroll
        for (int i = 0; i < 4; ++i)
            af[i] = wm_load_a16(&sA[cur][(64 * wr + 16 * i) * LDT], LDT, lane);
        v16h bf[2];
#pragma unroll
        for (int j = 0; j < 2; ++j)
            bf[j] = wm_load_bT16(&sW[cur][(32 * wc + 16 * j) * LDT], LDT, lane);
#pragma unroll
        for (int i = 0; i < 4; ++i)
#pragma unroll
            for (int j = 0; j < 2; ++j) acc[i][j] = wm_mma(af[i], bf[j], acc[i][j]);

        async_wait0();             // ...drained only after the WMMA block
        __syncthreads();
    }

    // Epilogue: bias, activation, optional row roll, store f16
    const int nloc = lane & 15;
    const int hi   = (lane >> 4) & 1;
#pragma unroll
    for (int mt = 0; mt < 4; ++mt) {
#pragma unroll
        for (int nt = 0; nt < 2; ++nt) {
            int col  = n0 + 32 * wc + 16 * nt + nloc;
            float bv = bias[col];
#pragma unroll
            for (int r = 0; r < 8; ++r) {
                int mrow = m0 + 64 * wr + 16 * mt + r + hi * 8;
                float v  = acc[mt][nt][r] + bv;
                if (ACT == 1) v = v / (1.0f + expf(-v));    // SiLU
                int b = mrow >> 14, t = mrow & (L_ - 1);
                int orow = (b << 14) | ((t + shift) & (L_ - 1));
                Out[(size_t)orow * N + col] = (_Float16)v;
            }
        }
    }
}

// ---------------------------------------------------------------------------
// Attention: one block per (window, head). 128 threads = 4 waves.
// Y layout: [32768][1536] f16, cols 0..511 = q, 512..1023 = k, 1024..1535 = v.
// ---------------------------------------------------------------------------
#define LDQ 40
#define LDV 136
#define LDS_ 132
#define ATTN_SMEM (2 * 128 * LDQ * 2 + 32 * LDV * 2 + 2 * 128 * 4 + 128 * LDS_ * 4)

__global__ __launch_bounds__(128) void attn_kernel(const _Float16* __restrict__ Y,
                                                   const float* __restrict__ lscale,
                                                   const float* __restrict__ biasT,
                                                   _Float16* __restrict__ Out) {
    extern __shared__ char smem[];
    _Float16* qs = (_Float16*)smem;                 // 128 x LDQ
    _Float16* ks = qs + 128 * LDQ;                  // 128 x LDQ
    _Float16* vt = ks + 128 * LDQ;                  // 32 x LDV (transposed V)
    float* rq = (float*)(vt + 32 * LDV);            // 128
    float* rk = rq + 128;                           // 128
    float* S  = rk + 128;                           // 128 x LDS_ (fp32 scores)

    const int wh   = blockIdx.x;
    const int head = wh & (NH_ - 1);
    const int win  = wh >> 4;                       // 0..255 (b*128 + wloc)
    const int wloc = win & (NW_ - 1);
    const size_t rowbase = (size_t)win * WSZ;
    const int tid = threadIdx.x;

    // Stage q, k (async DMA) and transposed v (VALU path) into LDS
    {
        const _Float16* q0 = Y + (rowbase + tid) * (3 * C_) + head * HD_;
#pragma unroll
        for (int j = 0; j < 4; ++j)
            async_copy_b128(q0 + j * 8, &qs[tid * LDQ + j * 8]);
        const _Float16* k0 = q0 + C_;
#pragma unroll
        for (int j = 0; j < 4; ++j)
            async_copy_b128(k0 + j * 8, &ks[tid * LDQ + j * 8]);
        const _Float16* v0 = q0 + 2 * C_;
#pragma unroll
        for (int j = 0; j < 4; ++j) {
            v8h vv = *(const v8h*)(v0 + j * 8);
#pragma unroll
            for (int e = 0; e < 8; ++e) vt[(j * 8 + e) * LDV + tid] = vv[e];
        }
    }
    async_wait0();
    __syncthreads();

    // Row L2-norm reciprocals (cosine attention folded into S epilogue)
    {
        float sq = 0.0f, sk = 0.0f;
#pragma unroll
        for (int d = 0; d < HD_; ++d) {
            float a = (float)qs[tid * LDQ + d]; sq += a * a;
            float b = (float)ks[tid * LDQ + d]; sk += b * b;
        }
        rq[tid] = 1.0f / fmaxf(sqrtf(sq), 1e-12f);
        rk[tid] = 1.0f / fmaxf(sqrtf(sk), 1e-12f);
    }
    const float sc = expf(fminf(lscale[head], LOGIT_MAX));
    __syncthreads();

    const int wave = tid >> 5, lane = tid & 31;
    const int nloc = lane & 15, hi = (lane >> 4) & 1;

    // S = (q @ k^T) * scale/(|q||k|) + cpb + mask   (each wave: 32 rows)
    {
        v16h a0 = wm_load_a16(&qs[(32 * wave) * LDQ], LDQ, lane);
        v16h a1 = wm_load_a16(&qs[(32 * wave + 16) * LDQ], LDQ, lane);
#pragma unroll
        for (int nt = 0; nt < 8; ++nt) {
            v16h b  = wm_load_bT16(&ks[(16 * nt) * LDQ], LDQ, lane);
            v8f c0 = wm_mma(a0, b, wm_zero());
            v8f c1 = wm_mma(a1, b, wm_zero());
            int   j   = 16 * nt + nloc;
            float rkj = rk[j] * sc;
            bool  jlo = (j < SSZ);
#pragma unroll
            for (int r = 0; r < 8; ++r) {
                int i0 = 32 * wave + r + hi * 8;
                float v0 = c0[r] * rq[i0] * rkj + biasT[(i0 - j + WSZ - 1) * NH_ + head];
                if (wloc == NW_ - 1 && ((i0 < SSZ) != jlo)) v0 -= 100.0f;
                S[i0 * LDS_ + j] = v0;
                int i1 = i0 + 16;
                float v1 = c1[r] * rq[i1] * rkj + biasT[(i1 - j + WSZ - 1) * NH_ + head];
                if (wloc == NW_ - 1 && ((i1 < SSZ) != jlo)) v1 -= 100.0f;
                S[i1 * LDS_ + j] = v1;
            }
        }
    }
    __syncthreads();

    // In-place softmax over each row (thread t owns row t)
    {
        float* row = S + tid * LDS_;
        float mx = -1e30f;
        for (int j = 0; j < WSZ; ++j) mx = fmaxf(mx, row[j]);
        float sum = 0.0f;
        for (int j = 0; j < WSZ; ++j) { float e = expf(row[j] - mx); row[j] = e; sum += e; }
        float inv = 1.0f / sum;
        for (int j = 0; j < WSZ; ++j) row[j] *= inv;
    }
    __syncthreads();

    // O = P @ V  (M=32 per wave, N=32, K=128)
    {
        v8f acc[2][2];
#pragma unroll
        for (int i = 0; i < 2; ++i)
#pragma unroll
            for (int j = 0; j < 2; ++j) acc[i][j] = wm_zero();
#pragma unroll
        for (int kc = 0; kc < 4; ++kc) {
            v16h a0 = wm_load_a32(&S[(32 * wave) * LDS_ + kc * 32], LDS_, lane);
            v16h a1 = wm_load_a32(&S[(32 * wave + 16) * LDS_ + kc * 32], LDS_, lane);
            v16h b0 = wm_load_bT16(&vt[0 * LDV + kc * 32], LDV, lane);
            v16h b1 = wm_load_bT16(&vt[16 * LDV + kc * 32], LDV, lane);
            acc[0][0] = wm_mma(a0, b0, acc[0][0]);
            acc[0][1] = wm_mma(a0, b1, acc[0][1]);
            acc[1][0] = wm_mma(a1, b0, acc[1][0]);
            acc[1][1] = wm_mma(a1, b1, acc[1][1]);
        }
#pragma unroll
        for (int mt = 0; mt < 2; ++mt)
#pragma unroll
            for (int nt = 0; nt < 2; ++nt) {
                int col = head * HD_ + 16 * nt + nloc;
#pragma unroll
                for (int r = 0; r < 8; ++r) {
                    int i = 32 * wave + 16 * mt + r + hi * 8;
                    Out[(rowbase + i) * C_ + col] = (_Float16)acc[mt][nt][r];
                }
            }
    }
}

// ---------------------------------------------------------------------------
// Residual + RMS-norm:  out = xin + y/rms(y) * w * sqrt(C)
// One wave (32 lanes) per token row of 512; 8 tokens per 256-thread block.
// ---------------------------------------------------------------------------
template <bool WRITE_H>
__global__ __launch_bounds__(256) void resid_rms(const float* __restrict__ xin,
                                                 const _Float16* __restrict__ y,
                                                 const float* __restrict__ w,
                                                 float* __restrict__ xf,
                                                 _Float16* __restrict__ xh) {
    int token = blockIdx.x * 8 + (threadIdx.x >> 5);
    int lane  = threadIdx.x & 31;
    const _Float16* yr = y + (size_t)token * C_ + lane * 16;
    v8h y0 = *(const v8h*)yr;
    v8h y1 = *(const v8h*)(yr + 8);
    float v[16], s = 0.0f;
#pragma unroll
    for (int i = 0; i < 8; ++i) { v[i] = (float)y0[i]; v[8 + i] = (float)y1[i]; }
#pragma unroll
    for (int i = 0; i < 16; ++i) s += v[i] * v[i];
#pragma unroll
    for (int off = 16; off > 0; off >>= 1) s += __shfl_xor(s, off, 32);
    float inv = rsqrtf(s * (1.0f / C_) + 1e-8f) * 22.627416997969522f;  // sqrt(512)/rms
    const float* xr = xin + (size_t)token * C_ + lane * 16;
    size_t o = (size_t)token * C_ + lane * 16;
#pragma unroll
    for (int i = 0; i < 16; ++i) {
        float r = xr[i] + v[i] * inv * w[lane * 16 + i];
        xf[o + i] = r;
        if (WRITE_H) xh[o + i] = (_Float16)r;
    }
}

// ---------------------------------------------------------------------------
// Launch
// ---------------------------------------------------------------------------
extern "C" void kernel_launch(void* const* d_in, const int* in_sizes, int n_in,
                              void* d_out, int out_size, void* d_ws, size_t ws_size,
                              hipStream_t stream) {
    (void)in_sizes; (void)n_in; (void)out_size; (void)ws_size;
    const float* x        = (const float*)d_in[0];
    const float* qkv_w    = (const float*)d_in[1];
    const float* q_bias   = (const float*)d_in[2];
    const float* v_bias   = (const float*)d_in[3];
    const float* lscale   = (const float*)d_in[4];
    const float* cpb_w1   = (const float*)d_in[5];
    const float* cpb_b1   = (const float*)d_in[6];
    const float* cpb_w2   = (const float*)d_in[7];
    const float* proj_w   = (const float*)d_in[8];
    const float* proj_b   = (const float*)d_in[9];
    const float* norm1_w  = (const float*)d_in[10];
    const float* norm2_w  = (const float*)d_in[11];
    const float* mlp_w1   = (const float*)d_in[12];
    const float* mlp_b1   = (const float*)d_in[13];
    const float* mlp_w2   = (const float*)d_in[14];
    const float* mlp_b2   = (const float*)d_in[15];

    char* ws = (char*)d_ws;
    const size_t MB = 1ull << 20;
    _Float16* Y   = (_Float16*)(ws + 0);          //  96 MB  qkv out (reused for H)
    _Float16* ATT = (_Float16*)(ws + 128 * MB);   //  32 MB  attention out
    _Float16* XS  = (_Float16*)(ws + 160 * MB);   //  32 MB  shifted x f16
    _Float16* PRJ = XS;                           //  reuse: proj out (XS dead)
    float*    X1F = (float*)   (ws + 192 * MB);   //  64 MB  x1 fp32
    _Float16* X1H = (_Float16*)(ws + 256 * MB);   //  32 MB  x1 f16
    _Float16* H   = (_Float16*)(ws + 0);          // 128 MB  mlp hidden (Y dead)
    _Float16* Y2  = (_Float16*)(ws + 128 * MB);   //  32 MB  mlp out (ATT dead)
    float*    QB  = (float*)   (ws + 288 * MB);
    float*    BT  = (float*)   (ws + 288 * MB + 64 * 1024);

    // Prep
    prep_qkvb<<<6, 256, 0, stream>>>(q_bias, v_bias, QB);
    prep_cpb<<<2 * WSZ - 1, 256, 0, stream>>>(cpb_w1, cpb_b1, cpb_w2, BT);
    prep_shift_cast<<<(MTOK * C_) / (256 * 8), 256, 0, stream>>>(x, XS);

    // QKV GEMM: [32768,512] x [1536,512]^T
    gemm_f16<0><<<dim3(12, MTOK / 128), 256, 0, stream>>>(XS, qkv_w, QB, Y,
                                                          3 * C_, C_, 0);

    // Attention (per window x head), large dynamic LDS
    hipFuncSetAttribute((const void*)attn_kernel,
                        hipFuncAttributeMaxDynamicSharedMemorySize, ATTN_SMEM);
    attn_kernel<<<B_ * NW_ * NH_, 128, ATTN_SMEM, stream>>>(Y, lscale, BT, ATT);

    // Projection (stores rows rolled back by +SS)
    gemm_f16<0><<<dim3(4, MTOK / 128), 256, 0, stream>>>(ATT, proj_w, proj_b, PRJ,
                                                         C_, C_, SSZ);

    // x1 = x + rmsnorm(proj_out)
    resid_rms<true><<<MTOK / 8, 256, 0, stream>>>(x, PRJ, norm1_w, X1F, X1H);

    // MLP: H = silu(x1 @ W1^T + b1); y2 = H @ W2^T + b2
    gemm_f16<1><<<dim3(16, MTOK / 128), 256, 0, stream>>>(X1H, mlp_w1, mlp_b1, H,
                                                          HID_, C_, 0);
    gemm_f16<0><<<dim3(4, MTOK / 128), 256, 0, stream>>>(H, mlp_w2, mlp_b2, Y2,
                                                         C_, HID_, 0);

    // out = x1 + rmsnorm(y2)
    resid_rms<false><<<MTOK / 8, 256, 0, stream>>>(X1F, Y2, norm2_w, (float*)d_out,
                                                   nullptr);
}
